// GCN3D_Feb14_PoolingDeep_31293131719377
// MI455X (gfx1250) — matrix operand order repaired and checked by reading, not compiled, not run on hardware
//
#include <hip/hip_runtime.h>

#define NNODES 100000
#define NCL 1024
#define EPSV 1e-5f

typedef __attribute__((ext_vector_type(16))) __bf16 v16bf;
typedef __attribute__((ext_vector_type(8)))  float v8f;

struct __attribute__((aligned(16))) U128 { unsigned int a, b, c, d; };
union Frag16 { U128 u[2]; v16bf v; };

#define BROW 40   // LDS row stride in shorts (80B: 16B-aligned, bank-conflict-free)

__device__ __forceinline__ unsigned short f2bf(float f) {
  unsigned int u = __float_as_uint(f);
  u = u + 0x7FFFu + ((u >> 16) & 1u);
  return (unsigned short)(u >> 16);
}

__device__ __forceinline__ float eluf(float v) {
  return v > 0.0f ? v : (__expf(v) - 1.0f);
}

// ---------------- conversion kernels ----------------

__global__ void k_act_to_bf16(const float* __restrict__ in, unsigned short* __restrict__ out,
                              long M, int K, int Kpad) {
  long idx = (long)blockIdx.x * blockDim.x + threadIdx.x;
  if (idx >= M * (long)Kpad) return;
  int k = (int)(idx % Kpad);
  long r = idx / Kpad;
  float v = (k < K) ? in[r * K + k] : 0.0f;
  out[idx] = f2bf(v);
}

__global__ void k_w_to_bf16t(const float* __restrict__ W, unsigned short* __restrict__ Wt,
                             int K, int Nact, int Kpad, int Npad16) {
  int idx = blockIdx.x * blockDim.x + threadIdx.x;
  if (idx >= Npad16 * Kpad) return;
  int k = idx % Kpad;
  int n = idx / Kpad;
  float v = (k < K && n < Nact) ? W[(size_t)k * Nact + n] : 0.0f;
  Wt[idx] = f2bf(v);
}

// ---------------- WMMA GEMM ----------------
// out[M,Nact] = A[M,K] @ W[K,Nact] (+bias, +elu).
// A: bf16 [M,Kpad]. Wt: bf16 W^T [Npad16,Kpad].
// Block = 8 waves; wave w -> 32 rows (two 16-row tiles), NT col-tiles (16*NT cols).
// B tile (NT*16 rows x 32 K) staged in double-buffered LDS, shared by all 8 waves;
// A fragments double-buffered in registers. Requires M % 32 == 0.

template <int NT>
__global__ __launch_bounds__(256) void k_wmma_gemm(
    const unsigned short* __restrict__ A,
    const unsigned short* __restrict__ Wt,
    const float* __restrict__ bias,
    float* __restrict__ out,
    int M, int Nact, int Kpad, int nBase0, int act) {
  __shared__ unsigned short Bs[2][64 * BROW];

  const int tid   = threadIdx.x;
  // force wave id into an SGPR so `active` is a scalar branch (EXEC untouched)
  const int wave  = __builtin_amdgcn_readfirstlane(tid) >> 5;
  const int lane  = tid & 31;
  const int hi    = lane >> 4;     // 0: lanes 0-15, 1: lanes 16-31
  const int lrow  = lane & 15;
  const int nBase = nBase0 + blockIdx.y * 64;
  const int mTile = blockIdx.x * 8 + wave;       // 32-row tile index
  const bool active = mTile < (M >> 5);

  const unsigned short* arow0 = A + (size_t)(mTile * 32 + lrow) * Kpad + hi * 8;
  const unsigned short* arow1 = arow0 + (size_t)16 * Kpad;

  // cooperative B-load mapping: thread -> (row, 16B chunk)
  const int br = tid >> 2;          // 0..63
  const int bc = (tid & 3) * 8;     // short offset within row
  const unsigned short* wrow = Wt + (size_t)(nBase + br) * Kpad + bc;

  const v8f vzero = {0.f, 0.f, 0.f, 0.f, 0.f, 0.f, 0.f, 0.f};
  v8f acc0[NT], acc1[NT];
#pragma unroll
  for (int t = 0; t < NT; ++t) { acc0[t] = vzero; acc1[t] = vzero; }

  // preload first A fragments + first B tile
  Frag16 aCur0, aCur1, aNxt0, aNxt1;
  if (active) {
    aCur0.u[0] = *(const U128*)(arow0);
    aCur0.u[1] = *(const U128*)(arow0 + 16);
    aCur1.u[0] = *(const U128*)(arow1);
    aCur1.u[1] = *(const U128*)(arow1 + 16);
  }
  if (br < NT * 16) *(U128*)(&Bs[0][br * BROW + bc]) = *(const U128*)(wrow);
  __syncthreads();

  int buf = 0;
  for (int k0 = 0; k0 < Kpad; k0 += 32) {
    const int k1 = k0 + 32;
    if (k1 < Kpad) {
      if (br < NT * 16) *(U128*)(&Bs[buf ^ 1][br * BROW + bc]) = *(const U128*)(wrow + k1);
      if (active) {
        aNxt0.u[0] = *(const U128*)(arow0 + k1);
        aNxt0.u[1] = *(const U128*)(arow0 + k1 + 16);
        aNxt1.u[0] = *(const U128*)(arow1 + k1);
        aNxt1.u[1] = *(const U128*)(arow1 + k1 + 16);
        __builtin_prefetch(arow0 + k1 + 32, 0, 1);
      }
    }
    if (active) {
      // load all B fragments first (one ds clause, one wait), then wmma back-to-back
      Frag16 bfr[NT];
#pragma unroll
      for (int t = 0; t < NT; ++t) {
        const unsigned short* bp = &Bs[buf][(t * 16 + lrow) * BROW + hi * 8];
        bfr[t].u[0] = *(const U128*)(bp);
        bfr[t].u[1] = *(const U128*)(bp + 16);
      }
#pragma unroll
      for (int t = 0; t < NT; ++t)
        acc0[t] = __builtin_amdgcn_wmma_f32_16x16x32_bf16(
            false, aCur0.v, false, bfr[t].v, (short)0, acc0[t], false, false);
#pragma unroll
      for (int t = 0; t < NT; ++t)
        acc1[t] = __builtin_amdgcn_wmma_f32_16x16x32_bf16(
            false, aCur1.v, false, bfr[t].v, (short)0, acc1[t], false, false);
    }
    __syncthreads();
    aCur0 = aNxt0; aCur1 = aNxt1;
    buf ^= 1;
  }

  if (!active) return;
#pragma unroll
  for (int t = 0; t < NT; ++t) {
    int n = nBase + t * 16 + lrow;
    if (n >= Nact) continue;
    float bv = bias ? bias[n] : 0.0f;
#pragma unroll
    for (int r = 0; r < 8; ++r) {
      int m0 = mTile * 32 + hi * 8 + r;
      float v0 = acc0[t][r] + bv;
      float v1 = acc1[t][r] + bv;
      if (act) { v0 = eluf(v0); v1 = eluf(v1); }
      out[(size_t)m0 * Nact + n] = v0;
      out[(size_t)(m0 + 16) * Nact + n] = v1;
    }
  }
}

// ---------------- degree / norm ----------------

__global__ void k_edge_deg(const int* __restrict__ dst, float* __restrict__ deg, int E) {
  int e = blockIdx.x * blockDim.x + threadIdx.x;
  if (e >= E) return;
  atomicAdd(&deg[dst[e]], 1.0f);
}

__global__ void k_dinv(const float* __restrict__ deg, float* __restrict__ dinv, int N) {
  int i = blockIdx.x * blockDim.x + threadIdx.x;
  if (i >= N) return;
  dinv[i] = rsqrtf(deg[i] + 1.0f);  // +1 self loop
}

// out = h * dinv^2 (self-loop message init)
__global__ void k_selfinit(const float* __restrict__ h, const float* __restrict__ dinv,
                           float* __restrict__ out, long N, int F) {
  long idx = (long)blockIdx.x * blockDim.x + threadIdx.x;
  if (idx >= N * (long)F) return;
  long i = idx / F;
  float d = dinv[i];
  out[idx] = h[idx] * d * d;
}

// warp per edge: out[dst] += h[src] * dinv[src]*dinv[dst]
__global__ void k_edge_agg(const float* __restrict__ h, const int* __restrict__ src,
                           const int* __restrict__ dst, const float* __restrict__ dinv,
                           float* __restrict__ out, int E, int F) {
  int gid  = blockIdx.x * blockDim.x + threadIdx.x;
  int wid  = gid >> 5;
  int lane = gid & 31;
  if (wid >= E) return;
  int s = src[wid], d = dst[wid];
  float c = dinv[s] * dinv[d];
  const float* hs = h + (size_t)s * F;
  float* od = out + (size_t)d * F;
  for (int f = lane; f < F; f += 32) atomicAdd(&od[f], hs[f] * c);
}

__global__ void k_bias_elu(float* __restrict__ buf, const float* __restrict__ bias,
                           long N, int F) {
  long idx = (long)blockIdx.x * blockDim.x + threadIdx.x;
  if (idx >= N * (long)F) return;
  int f = (int)(idx % F);
  buf[idx] = eluf(buf[idx] + bias[f]);
}

// ---------------- instance norm ----------------

__global__ void k_in_stats(const float* __restrict__ y, float* __restrict__ stat, int N) {
  int f = threadIdx.x;  // 256 threads = 256 features
  float s = 0.f, ss = 0.f;
  for (int i = blockIdx.x; i < N; i += gridDim.x) {
    float v = y[(size_t)i * 256 + f];
    s += v; ss += v * v;
  }
  atomicAdd(&stat[f], s);
  atomicAdd(&stat[256 + f], ss);
}

__global__ void k_in_apply(const float* __restrict__ yin, const float* __restrict__ stat,
                           float* __restrict__ yout, long N) {
  long idx = (long)blockIdx.x * blockDim.x + threadIdx.x;
  if (idx >= N * 256) return;
  int f = (int)(idx & 255);
  float inv = 1.0f / (float)N;
  float mu = stat[f] * inv;
  float var = stat[256 + f] * inv - mu * mu;
  yout[idx] = (yin[idx] - mu) * rsqrtf(var + EPSV);
}

// ---------------- cluster pooling ----------------

__global__ void k_pool_accum(const float* __restrict__ y, const int* __restrict__ cl,
                             const int* __restrict__ ib, float* __restrict__ cnt,
                             float* __restrict__ psum, int N) {
  int gid  = blockIdx.x * blockDim.x + threadIdx.x;
  int wid  = gid >> 5;
  int lane = gid & 31;
  if (wid >= N) return;
  int c = cl[wid] + ib[wid] * NCL;
  if (lane == 0) atomicAdd(&cnt[c], 1.0f);
  const float* yr = y + (size_t)wid * 256;
  float* pr = psum + (size_t)c * 256;
  for (int f = lane; f < 256; f += 32) atomicAdd(&pr[f], yr[f]);
}

__global__ void k_pool_div(const float* __restrict__ psum, const float* __restrict__ cnt,
                           float* __restrict__ px) {
  int idx = blockIdx.x * blockDim.x + threadIdx.x;
  if (idx >= NCL * 256) return;
  int c = idx >> 8;
  float k = cnt[c];
  px[idx] = psum[idx] / (k > 1.0f ? k : 1.0f);
}

// unique (psrc,pdst) pairs, no self loops == sort+first-mask dedupe in the reference
__global__ void k_mask(const int* __restrict__ src, const int* __restrict__ dst,
                       const int* __restrict__ cl, const int* __restrict__ ib,
                       int* __restrict__ mask, int E) {
  int e = blockIdx.x * blockDim.x + threadIdx.x;
  if (e >= E) return;
  int s = src[e], d = dst[e];
  int ps = cl[s] + ib[s] * NCL;
  int pd = cl[d] + ib[d] * NCL;
  if (ps != pd) mask[ps * NCL + pd] = 1;
}

__global__ void k_local_dinv(const int* __restrict__ mask, float* __restrict__ dinvL) {
  int d = blockIdx.x * blockDim.x + threadIdx.x;
  if (d >= NCL) return;
  float deg = 0.f;
  for (int s = 0; s < NCL; ++s) deg += (float)mask[s * NCL + d];
  dinvL[d] = rsqrtf(deg + 1.0f);
}

// dense masked aggregation on the 1024-cluster graph
__global__ void k_local_agg(const float* __restrict__ h, const int* __restrict__ mask,
                            const float* __restrict__ dinvL, float* __restrict__ out, int F) {
  int d = blockIdx.x;  // 1024 blocks
  float dd = dinvL[d];
  for (int f = threadIdx.x; f < F; f += blockDim.x) {
    float acc = 0.f;
    for (int s = 0; s < NCL; ++s) {
      if (mask[s * NCL + d]) acc += h[(size_t)s * F + f] * dinvL[s];
    }
    out[(size_t)d * F + f] = acc * dd + h[(size_t)d * F + f] * dd * dd;
  }
}

// ---------------- concat y[.,256] || comp[bc,64] ----------------

__global__ void k_concat(const float* __restrict__ Y, const float* __restrict__ comp,
                         const int* __restrict__ cl, const int* __restrict__ ib,
                         float* __restrict__ z, long N) {
  long idx = (long)blockIdx.x * blockDim.x + threadIdx.x;
  if (idx >= N * 320) return;
  long i = idx / 320;
  int c = (int)(idx % 320);
  float v;
  if (c < 256) v = Y[i * 256 + c];
  else {
    int b = cl[i] + ib[i] * NCL;
    v = comp[(size_t)b * 64 + (c - 256)];
  }
  z[idx] = v;
}

// ---------------- host side ----------------

static inline int cdiv(long a, long b) { return (int)((a + b - 1) / b); }

static void launch_gemm(const float* Xf, int M, int K,
                        const float* W, const float* bias, int Nact, int act,
                        unsigned short* Abf, unsigned short* Wt, float* out,
                        hipStream_t s) {
  int Kpad = cdiv(K, 32) * 32;
  int Npad16 = cdiv(Nact, 16) * 16;
  long atot = (long)M * Kpad;
  k_act_to_bf16<<<cdiv(atot, 256), 256, 0, s>>>(Xf, Abf, (long)M, K, Kpad);
  k_w_to_bf16t<<<cdiv((long)Npad16 * Kpad, 256), 256, 0, s>>>(W, Wt, K, Nact, Kpad, Npad16);

  int gx = cdiv(M, 256);   // 8 waves x 32 rows per block
  int full = Npad16 / 64;
  int rem  = (Npad16 % 64) / 16;
  if (full > 0)
    k_wmma_gemm<4><<<dim3(gx, full), 256, 0, s>>>(Abf, Wt, bias, out, M, Nact, Kpad, 0, act);
  if (rem == 1)
    k_wmma_gemm<1><<<dim3(gx, 1), 256, 0, s>>>(Abf, Wt, bias, out, M, Nact, Kpad, full * 64, act);
  else if (rem == 2)
    k_wmma_gemm<2><<<dim3(gx, 1), 256, 0, s>>>(Abf, Wt, bias, out, M, Nact, Kpad, full * 64, act);
  else if (rem == 3)
    k_wmma_gemm<3><<<dim3(gx, 1), 256, 0, s>>>(Abf, Wt, bias, out, M, Nact, Kpad, full * 64, act);
}

extern "C" void kernel_launch(void* const* d_in, const int* in_sizes, int n_in,
                              void* d_out, int out_size, void* d_ws, size_t ws_size,
                              hipStream_t stream) {
  const int N = in_sizes[0] / 3;       // 100000
  const int E = in_sizes[1] / 2;       // 1600000

  const float* x   = (const float*)d_in[0];
  const int* adj   = (const int*)d_in[1];
  const int* src   = adj;
  const int* dst   = adj + E;
  const int* ib    = (const int*)d_in[3];
  const int* cl    = (const int*)d_in[4];

  const float *WG1=(const float*)d_in[5],  *bG1=(const float*)d_in[6];
  const float *WfG1=(const float*)d_in[7], *bfG1=(const float*)d_in[8];
  const float *WG2=(const float*)d_in[9],  *bG2=(const float*)d_in[10];
  const float *WfG2=(const float*)d_in[11],*bfG2=(const float*)d_in[12];
  const float *WL1=(const float*)d_in[13], *bL1=(const float*)d_in[14];
  const float *WfL1=(const float*)d_in[15],*bfL1=(const float*)d_in[16];
  const float *WL2=(const float*)d_in[17], *bL2=(const float*)d_in[18];
  const float *WfL2=(const float*)d_in[19],*bfL2=(const float*)d_in[20];
  const float *WO1=(const float*)d_in[21], *bO1=(const float*)d_in[22];
  const float *WfO1=(const float*)d_in[23],*bfO1=(const float*)d_in[24];
  const float *WO2=(const float*)d_in[25], *bO2=(const float*)d_in[26];
  const float *WfO2=(const float*)d_in[27],*bfO2=(const float*)d_in[28];
  const float *WO3=(const float*)d_in[29], *bO3=(const float*)d_in[30];
  const float *WfO3=(const float*)d_in[31],*bfO3=(const float*)d_in[32];

  // workspace carve-up
  size_t off = 0;
  char* base = (char*)d_ws;
  auto carve = [&](size_t bytes) -> char* {
    char* p = base + off;
    off += (bytes + 255) & ~(size_t)255;
    return p;
  };
  float* Y    = (float*)carve((size_t)N * 256 * 4);
  float* bufA = (float*)carve((size_t)N * 320 * 4);
  float* bufB = (float*)carve((size_t)N * 320 * 4);
  float* bufC = (float*)carve((size_t)N * 256 * 4);
  unsigned short* Abf  = (unsigned short*)carve((size_t)N * 320 * 2);
  unsigned short* Wt   = (unsigned short*)carve(256 * 1024);
  float* deg  = (float*)carve((size_t)N * 4);
  float* dinv = (float*)carve((size_t)N * 4);
  float* stat = (float*)carve(512 * 4);
  float* cnt  = (float*)carve(NCL * 4);
  float* psum = (float*)carve((size_t)NCL * 256 * 4);
  float* px   = (float*)carve((size_t)NCL * 256 * 4);
  int*   msk  = (int*)carve((size_t)NCL * NCL * 4);
  float* dinvL= (float*)carve(NCL * 4);
  float* lA   = (float*)carve((size_t)NCL * 256 * 4);
  float* lB   = (float*)carve((size_t)NCL * 256 * 4);
  unsigned short* lAbf = (unsigned short*)carve((size_t)NCL * 320 * 2);

  // ---- node-graph symmetric norm (shared by all node convs, ew = 1) ----
  hipMemsetAsync(deg, 0, (size_t)N * 4, stream);
  k_edge_deg<<<cdiv(E, 256), 256, 0, stream>>>(dst, deg, E);
  k_dinv<<<cdiv(N, 256), 256, 0, stream>>>(deg, dinv, N);

  auto node_gcn = [&](const float* xin, int K, const float* W, const float* b, int F,
                      float* tbuf, float* obuf) {
    launch_gemm(xin, N, K, W, nullptr, F, /*act*/0, Abf, Wt, tbuf, stream);
    k_selfinit<<<cdiv((long)N * F, 256), 256, 0, stream>>>(tbuf, dinv, obuf, N, F);
    k_edge_agg<<<cdiv((long)E * 32, 256), 256, 0, stream>>>(tbuf, src, dst, dinv, obuf, E, F);
    k_bias_elu<<<cdiv((long)N * F, 256), 256, 0, stream>>>(obuf, b, (long)N, F);
  };
  auto local_gcn = [&](const float* xin, int K, const float* W, const float* b, int F,
                       float* tbuf, float* obuf) {
    launch_gemm(xin, NCL, K, W, nullptr, F, 0, lAbf, Wt, tbuf, stream);
    k_local_agg<<<NCL, 128, 0, stream>>>(tbuf, msk, dinvL, obuf, F);
    k_bias_elu<<<cdiv((long)NCL * F, 256), 256, 0, stream>>>(obuf, b, (long)NCL, F);
  };

  // ---- global GCN stack ----
  node_gcn(x, 3, WG1, bG1, 64, bufA, bufB);                                   // bufB [N,64]
  launch_gemm(bufB, N, 64, WfG1, bfG1, 64, 1, Abf, Wt, bufA, stream);         // bufA [N,64]
  node_gcn(bufA, 64, WG2, bG2, 256, bufB, bufC);                              // bufC [N,256]
  launch_gemm(bufC, N, 256, WfG2, bfG2, 256, 1, Abf, Wt, bufA, stream);       // bufA [N,256]

  // ---- instance norm -> Y ----
  hipMemsetAsync(stat, 0, 512 * 4, stream);
  k_in_stats<<<512, 256, 0, stream>>>(bufA, stat, N);
  k_in_apply<<<cdiv((long)N * 256, 256), 256, 0, stream>>>(bufA, stat, Y, (long)N);

  // ---- cluster avg-pool + pooled edge dedupe ----
  hipMemsetAsync(cnt, 0, NCL * 4, stream);
  hipMemsetAsync(psum, 0, (size_t)NCL * 256 * 4, stream);
  k_pool_accum<<<cdiv((long)N * 32, 256), 256, 0, stream>>>(Y, cl, ib, cnt, psum, N);
  k_pool_div<<<cdiv((long)NCL * 256, 256), 256, 0, stream>>>(psum, cnt, px);
  hipMemsetAsync(msk, 0, (size_t)NCL * NCL * 4, stream);
  k_mask<<<cdiv(E, 256), 256, 0, stream>>>(src, dst, cl, ib, msk, E);
  k_local_dinv<<<cdiv(NCL, 256), 256, 0, stream>>>(msk, dinvL);

  // ---- local GCN stack on compressed graph ----
  local_gcn(px, 256, WL1, bL1, 128, lA, lB);                                  // lB [C,128]
  launch_gemm(lB, NCL, 128, WfL1, bfL1, 128, 1, lAbf, Wt, lA, stream);        // lA [C,128]
  local_gcn(lA, 128, WL2, bL2, 64, lB, lA);                                   // lA [C,64]
  launch_gemm(lA, NCL, 64, WfL2, bfL2, 64, 1, lAbf, Wt, lB, stream);          // lB [C,64]

  // ---- concat and output stack ----
  k_concat<<<cdiv((long)N * 320, 256), 256, 0, stream>>>(Y, lB, cl, ib, bufB, (long)N); // z [N,320]
  node_gcn(bufB, 320, WO1, bO1, 128, bufA, bufC);                             // bufC [N,128]
  launch_gemm(bufC, N, 128, WfO1, bfO1, 128, 1, Abf, Wt, bufA, stream);       // bufA [N,128]
  node_gcn(bufA, 128, WO2, bO2, 32, bufB, bufC);                              // bufC [N,32]
  launch_gemm(bufC, N, 32, WfO2, bfO2, 32, 1, Abf, Wt, bufA, stream);         // bufA [N,32]
  node_gcn(bufA, 32, WO3, bO3, 3, bufB, bufC);                                // bufC [N,3]
  launch_gemm(bufC, N, 3, WfO3, bfO3, 3, 1, Abf, Wt, (float*)d_out, stream);  // out [N,3]
}
